// MHSA_9105330668017
// MI455X (gfx1250) — matrix operand order repaired and checked by reading, not compiled, not run on hardware
//
#include <hip/hip_runtime.h>
#include <hip/hip_bf16.h>

typedef _Float16 h16;
typedef __attribute__((ext_vector_type(8)))  _Float16 v8h;
typedef __attribute__((ext_vector_type(16))) _Float16 v16h;
typedef __attribute__((ext_vector_type(8)))  float    v8f;

constexpr int B_ = 4, S_ = 2048, D_ = 1024, H_ = 16, DH_ = 64;
constexpr float EPS_ = 1e-12f, MIN_TEMP_ = 0.01f;

union AFrag { v16h v; v8h h[2]; h16 e[16]; };

static __device__ inline v8f wmma_f16(v16h a, v16h b, v8f c) {
  return __builtin_amdgcn_wmma_f32_16x16x32_f16(false, a, false, b, (short)0, c,
                                                false, false);
}

static __device__ inline v8f vzero8f() {
  v8f z = {0.f, 0.f, 0.f, 0.f, 0.f, 0.f, 0.f, 0.f};
  return z;
}

// ---------------------------------------------------------------------------
// Kernel 1: per-head QKV projection via WMMA.
//   y^T-free formulation: A = x tile (16 seq rows x 64 ch, f32->f16 on load),
//   B = W^T staged once per block in LDS as f16 [d][e].
//   C layout keeps each output row inside one 16-lane half, so the L2-norm
//   reduction is 4 xor-shuffles per slot, once per 16 rows; the rescale is
//   already in slot space. temp clamp folded into q.
// Block: 256 threads (8 waves) covers (b, h, 128 seq rows).
// ---------------------------------------------------------------------------
__global__ __launch_bounds__(256) void qkv_proj_kernel(
    const float* __restrict__ x,
    const float* __restrict__ Wq, const float* __restrict__ bq,
    const float* __restrict__ Wk, const float* __restrict__ bk,
    const float* __restrict__ Wv, const float* __restrict__ bv,
    const float* __restrict__ temp,
    h16* __restrict__ qh, h16* __restrict__ kh, h16* __restrict__ vh) {
  __shared__ __attribute__((aligned(64))) h16 sWT[3][DH_ * DH_];  // [mat][d][e]
  __shared__ float sb[3][DH_];

  const int blk   = blockIdx.x;
  const int stile = blk % (S_ / 128);
  const int h     = (blk / (S_ / 128)) % H_;
  const int b     = blk / ((S_ / 128) * H_);
  const int t = threadIdx.x, wave = t >> 5, lane = t & 31;

  // stage transposed f16 weights (coalesced global reads)
  const int wbase = h * DH_ * DH_;
  for (int i = t; i < DH_ * DH_; i += 256) {
    const int e = i >> 6, d = i & 63;
    const int tr = d * DH_ + e;
    sWT[0][tr] = (h16)Wq[wbase + i];
    sWT[1][tr] = (h16)Wk[wbase + i];
    sWT[2][tr] = (h16)Wv[wbase + i];
  }
  if (t < DH_) {
    sb[0][t] = bq[h * DH_ + t];
    sb[1][t] = bk[h * DH_ + t];
    sb[2][t] = bv[h * DH_ + t];
  }
  __syncthreads();

  const float tscale = fmaxf(temp[h], MIN_TEMP_);
  const int l15 = lane & 15;
  const int hi  = lane >> 4;
  const int kb  = hi ? 8 : 0;
  const int s0  = stile * 128 + wave * 16;
  const size_t rowbase = ((size_t)b * H_ + h) * S_;

  // A fragments of the x tile (f32 -> f16 on load)
  AFrag aX[2];
  {
    const float* xrow = x + ((size_t)b * S_ + (s0 + l15)) * D_ + h * DH_;
#pragma unroll
    for (int f = 0; f < 2; ++f) {
      const float* xp = xrow + f * 32 + kb;
#pragma unroll
      for (int j = 0; j < 8; ++j) {
        aX[f].e[j]     = (h16)xp[j];
        aX[f].e[8 + j] = (h16)xp[16 + j];
      }
    }
  }

#pragma unroll
  for (int mat = 0; mat < 3; ++mat) {
    v8f c[4];
#pragma unroll
    for (int s = 0; s < 4; ++s) c[s] = vzero8f();
#pragma unroll
    for (int f = 0; f < 2; ++f) {
      const h16* wp = &sWT[mat][(f * 32 + kb) * DH_ + l15];
#pragma unroll
      for (int s = 0; s < 4; ++s) {
        AFrag bW;
#pragma unroll
        for (int j = 0; j < 8; ++j) {
          bW.e[j]     = wp[j * DH_ + s * 16];
          bW.e[8 + j] = wp[(16 + j) * DH_ + s * 16];
        }
        c[s] = wmma_f16(aX[f].v, bW.v, c[s]);
      }
    }
    // bias (f32)
#pragma unroll
    for (int s = 0; s < 4; ++s) {
      const float bb = sb[mat][s * 16 + l15];
#pragma unroll
      for (int r = 0; r < 8; ++r) c[s][r] += bb;
    }
    // per-row scale: L2 normalize for q/k (temp folded into q), identity for v
    float scl[8];
    if (mat < 2) {
      const float topn = (mat == 0) ? tscale : 1.0f;
#pragma unroll
      for (int r = 0; r < 8; ++r) {
        float ss = 0.0f;
#pragma unroll
        for (int s = 0; s < 4; ++s) ss += c[s][r] * c[s][r];
#pragma unroll
        for (int mask = 8; mask >= 1; mask >>= 1)
          ss += __shfl_xor(ss, mask, 32);
        scl[r] = topn / fmaxf(sqrtf(ss), EPS_);
      }
    } else {
#pragma unroll
      for (int r = 0; r < 8; ++r) scl[r] = 1.0f;
    }
    h16* op = (mat == 0) ? qh : (mat == 1) ? kh : vh;
#pragma unroll
    for (int r = 0; r < 8; ++r) {
      h16* orow = op + (rowbase + s0 + r + hi * 8) * DH_;
#pragma unroll
      for (int s = 0; s < 4; ++s)
        orow[s * 16 + l15] = (h16)(c[s][r] * scl[r]);
    }
  }
}

// ---------------------------------------------------------------------------
// Kernel 2: flash attention computing S^T = K * Q^T so that:
//   - K is the A operand: row-major, loaded straight from global (no staging)
//   - Q^T is the (loop-invariant) B operand: transposed once through LDS
//   - each query row's logits land in ONE lane -> softmax is per-lane VALU
//     (only one xor-16 shuffle to merge the two M-halves)
// Block: 256 threads (8 independent waves); wave owns 16 query rows.
// KV tile = 64 columns -> 16 WMMAs per iteration, 32 iterations.
// ---------------------------------------------------------------------------
__global__ __launch_bounds__(256) void attn_kernel(
    const h16* __restrict__ qh, const h16* __restrict__ kh,
    const h16* __restrict__ vh, float* __restrict__ out) {
  // per-wave scratch: first holds Q^T (64 dh x 16 q), then reused for P
  __shared__ __attribute__((aligned(64))) h16 sW[8][16 * 64];

  const int blk   = blockIdx.x;
  const int qtile = blk % (S_ / 128);
  const int h     = (blk / (S_ / 128)) % H_;
  const int b     = blk / ((S_ / 128) * H_);
  const int t = threadIdx.x, wave = t >> 5, lane = t & 31;

  const size_t base = ((size_t)b * H_ + h) * (size_t)S_ * DH_;
  const h16* Q = qh + base;
  const h16* K = kh + base;
  const h16* V = vh + base;

  const int l15 = lane & 15;
  const int hi  = lane >> 4;
  const int kb  = hi ? 8 : 0;
  const int q0  = qtile * 128 + wave * 16;
  h16* myW = &sW[wave][0];

  // ---- one-time: stage Q^T into per-wave LDS ([dh][qcol], row = 16 h16)
  {
    const h16* qp = Q + (size_t)(q0 + l15) * DH_ + hi * 32;
#pragma unroll
    for (int j = 0; j < 32; ++j) myW[(hi * 32 + j) * 16 + l15] = qp[j];
  }
  asm volatile("s_wait_dscnt 0" ::: "memory");
  v16h bQ[2];  // B-fragments: lane = q col (l15), elements = dh (table order)
#pragma unroll
  for (int f = 0; f < 2; ++f) {
    AFrag tq;
#pragma unroll
    for (int j = 0; j < 8; ++j) {
      tq.e[j]     = myW[(f * 32 + kb + j) * 16 + l15];
      tq.e[8 + j] = myW[(f * 32 + kb + 16 + j) * 16 + l15];
    }
    bQ[f] = tq.v;
  }

  float m = -3.0e38f, l = 0.0f;  // per-lane stats for query row q0 + l15
  v8f acc[4];
#pragma unroll
  for (int s = 0; s < 4; ++s) acc[s] = vzero8f();

#pragma unroll 1
  for (int kv0 = 0; kv0 < S_; kv0 += 64) {
    if (kv0 + 64 < S_) {  // prefetch next KV tile (global_prefetch_b8)
      __builtin_prefetch(K + (size_t)(kv0 + 64 + lane) * DH_, 0, 1);
      __builtin_prefetch(V + (size_t)(kv0 + 64 + lane) * DH_, 0, 1);
    }

    // ---- logits (transposed): c[tt] = K[16 kv rows] x Q^T, dh over 2 WMMAs
    v8f c[4];
#pragma unroll
    for (int tt = 0; tt < 4; ++tt) {
      const h16* kp = K + (size_t)(kv0 + tt * 16 + l15) * DH_ + kb;
      const v8h* p0 = (const v8h*)kp;
      const v8h* p1 = (const v8h*)(kp + 32);
      AFrag aK0, aK1;
      aK0.h[0] = p0[0]; aK0.h[1] = p0[2];  // dh = kb..kb+7, kb+16..kb+23
      aK1.h[0] = p1[0]; aK1.h[1] = p1[2];  // + 32
      c[tt] = wmma_f16(aK0.v, bQ[0], vzero8f());
      c[tt] = wmma_f16(aK1.v, bQ[1], c[tt]);
    }

    // ---- online softmax: all per-lane except one xor-16 half-merge
    float cmax = -3.0e38f;
#pragma unroll
    for (int tt = 0; tt < 4; ++tt)
#pragma unroll
      for (int r = 0; r < 8; ++r) cmax = fmaxf(cmax, c[tt][r]);
    cmax = fmaxf(cmax, __shfl_xor(cmax, 16, 32));
    const float mn = fmaxf(m, cmax);
    const float alpha = __expf(m - mn);
    m = mn;

    float p[4][8];
    float ls = 0.0f;
#pragma unroll
    for (int tt = 0; tt < 4; ++tt)
#pragma unroll
      for (int r = 0; r < 8; ++r) {
        p[tt][r] = __expf(c[tt][r] - mn);
        ls += p[tt][r];
      }
    ls += __shfl_xor(ls, 16, 32);
    l = l * alpha + ls;

    // broadcast alpha from lane-space (q = lane) into slot-space (q = slot)
    float asl[8];
#pragma unroll
    for (int r = 0; r < 8; ++r) {
      const float alo = __shfl(alpha, r, 32);
      const float ahi = __shfl(alpha, r + 8, 32);
      asl[r] = hi ? ahi : alo;
    }
#pragma unroll
    for (int s = 0; s < 4; ++s)
#pragma unroll
      for (int r = 0; r < 8; ++r) acc[s][r] *= asl[r];

    // ---- transpose P' (lane = q col, slot = kv) -> LDS [q][kv], packed b32
    {
      const int kvb = hi * 8;
#pragma unroll
      for (int tt = 0; tt < 4; ++tt)
#pragma unroll
        for (int j = 0; j < 4; ++j) {
          union { h16 hh[2]; unsigned u; } pk;
          pk.hh[0] = (h16)p[tt][2 * j];
          pk.hh[1] = (h16)p[tt][2 * j + 1];
          *(unsigned*)(myW + l15 * 64 + tt * 16 + kvb + 2 * j) = pk.u;
        }
    }
    asm volatile("s_wait_dscnt 0" ::: "memory");

    // ---- PV: A = P (16 q x 32 kv), B = V (lane = dh col, elements = kv)
#pragma unroll
    for (int g = 0; g < 2; ++g) {
      const v8h* pp = (const v8h*)(myW + l15 * 64 + g * 32 + kb);
      AFrag aP;
      aP.h[0] = pp[0];
      aP.h[1] = pp[2];
      const h16* vp = V + (size_t)(kv0 + g * 32 + kb) * DH_ + l15;
#pragma unroll
      for (int s = 0; s < 4; ++s) {
        AFrag vf;
#pragma unroll
        for (int j = 0; j < 8; ++j) {
          vf.e[j]     = vp[j * DH_ + s * 16];
          vf.e[8 + j] = vp[(16 + j) * DH_ + s * 16];
        }
        acc[s] = wmma_f16(aP.v, vf.v, acc[s]);
      }
    }
  }

  // ---- epilogue: 1/l broadcast into slot space, write f32 [B,S,H*DH]
  const float inv = 1.0f / l;  // valid for q row = q0 + l15
  float invs[8];
#pragma unroll
  for (int r = 0; r < 8; ++r) {
    const float ilo  = __shfl(inv, r, 32);
    const float ihiv = __shfl(inv, r + 8, 32);
    invs[r] = hi ? ihiv : ilo;
  }
#pragma unroll
  for (int r = 0; r < 8; ++r) {
    const int srow = q0 + r + hi * 8;  // C layout: M = slot + 8*hi
    float* op = out + (((size_t)b * S_ + srow) * H_ + h) * DH_;
#pragma unroll
    for (int s = 0; s < 4; ++s) op[s * 16 + l15] = acc[s][r] * invs[r];
  }
}

// ---------------------------------------------------------------------------
extern "C" void kernel_launch(void* const* d_in, const int* in_sizes, int n_in,
                              void* d_out, int out_size, void* d_ws,
                              size_t ws_size, hipStream_t stream) {
  (void)in_sizes; (void)n_in; (void)out_size; (void)ws_size;
  const float* x    = (const float*)d_in[0];
  const float* Wq   = (const float*)d_in[1];
  const float* bq   = (const float*)d_in[2];
  const float* Wk   = (const float*)d_in[3];
  const float* bk   = (const float*)d_in[4];
  const float* Wv   = (const float*)d_in[5];
  const float* bv   = (const float*)d_in[6];
  const float* temp = (const float*)d_in[7];
  float* out = (float*)d_out;

  const size_t N = (size_t)B_ * H_ * S_ * DH_;
  h16* qhp = (h16*)d_ws;
  h16* khp = qhp + N;
  h16* vhp = khp + N;

  qkv_proj_kernel<<<B_ * H_ * (S_ / 128), 256, 0, stream>>>(
      x, Wq, bq, Wk, bk, Wv, bv, temp, qhp, khp, vhp);
  attn_kernel<<<B_ * H_ * (S_ / 128), 256, 0, stream>>>(qhp, khp, vhp, out);
}
